// IndustrySparseAttention_1949915152747
// MI455X (gfx1250) — matrix operand order repaired and checked by reading, not compiled, not run on hardware
//
#include <hip/hip_runtime.h>
#include <hip/hip_bf16.h>

// ---------------------------------------------------------------------------
// IndustrySparseAttention for MI455X (gfx1250, wave32, WMMA f16 16x16x32).
// fp32->f16 convert; QKV GEMMs (bias fused, async double-buffered W tiles in
// LDS, 8 wmma per k-step per wave); flash attention (block-shared K/V tiles,
// async-to-LDS staging); output GEMM (fp32 out).
// ---------------------------------------------------------------------------

typedef __attribute__((ext_vector_type(16))) _Float16 v16h;
typedef __attribute__((ext_vector_type(8)))  float    v8f;
typedef __attribute__((ext_vector_type(4)))  int      v4i;

#define WMMA_F16(a, b, c) \
  __builtin_amdgcn_wmma_f32_16x16x32_f16(false, (a), false, (b), (short)0, (c), false, false)

#if defined(__has_builtin)
#if __has_builtin(__builtin_amdgcn_global_load_async_to_lds_b128)
#define HAVE_ASYNC_LDS 1
#endif
#if __has_builtin(__builtin_amdgcn_s_wait_asynccnt)
#define WAIT_ASYNCCNT(n) __builtin_amdgcn_s_wait_asynccnt(n)
#endif
#endif
#ifndef HAVE_ASYNC_LDS
#define HAVE_ASYNC_LDS 0
#endif
#ifndef WAIT_ASYNCCNT
#define WAIT_ASYNCCNT(n) asm volatile("s_wait_asynccnt %0" ::"i"(n) : "memory")
#endif

// Split workgroup barrier WITHOUT the full memory fence of __syncthreads()
// (no forced s_wait_loadcnt 0 -> global loads can pipeline across it).
#define BAR() asm volatile("s_barrier_signal -1\n\ts_barrier_wait -1" ::: "memory")
#define WAIT_DS0() asm volatile("s_wait_dscnt 0" ::: "memory")

// 16-byte global->LDS copy; async (ASYNCcnt-tracked) when available.
__device__ __forceinline__ void async_copy16(const _Float16* src, _Float16* dst) {
#if HAVE_ASYNC_LDS
  __builtin_amdgcn_global_load_async_to_lds_b128(
      (__attribute__((address_space(1))) v4i*)src,
      (__attribute__((address_space(3))) v4i*)dst, 0, 0);
#else
  *(uint4*)dst = *(const uint4*)src;
#endif
}

// B-fragment (32x16, 16-bit): lane L holds column N=L&15; element i <-> K = 16*(L>>4)+i.
__device__ __forceinline__ v16h load_bfrag_lds(const _Float16* p) {
  v16h r;
  ((uint4*)&r)[0] = *(const uint4*)(p);
  ((uint4*)&r)[1] = *(const uint4*)(p + 8);
  return r;
}

// A-fragment (16x32, 16-bit): lane L holds row M=L&15;
// element i <-> K = kbase + 16*(L>>4) + i.
__device__ __forceinline__ v16h load_afrag(const _Float16* rowp, int kbase, int lh) {
  v16h r;
  ((uint4*)&r)[0] = *(const uint4*)(rowp + kbase + 16 * lh);
  ((uint4*)&r)[1] = *(const uint4*)(rowp + kbase + 16 * lh + 8);
  return r;
}

// --------------------------- fp32 -> f16 convert ---------------------------
__global__ void cvt_f32_f16(const float* __restrict__ src, _Float16* __restrict__ dst, int n) {
  int i = blockIdx.x * 256 + threadIdx.x;
  if (i < n) dst[i] = (_Float16)src[i];
}

// ------------------------- shared GEMM tile core ---------------------------
// Per wave: 32x64 output tile (2 A-frags x 4 B-frags = 8 wmma / k-step).
// 256 threads (8 waves) -> block tile 256x64. W tile (64x32 halves = 4KB)
// double-buffered in LDS, staged 2 k-steps ahead; A-frags pipelined 1 ahead.
__device__ __forceinline__ void stage_W(const _Float16* W, _Float16* buf, int N0, int ks, int tid) {
  const int r = tid >> 2;          // 0..63
  const int c = (tid & 3) * 8;     // 0,8,16,24 halves
  async_copy16(W + (size_t)(N0 + r) * 512 + 32 * ks + c, buf + r * 32 + c);
}

__device__ __forceinline__ void gemm_tile(const _Float16* __restrict__ xh,
                                          const _Float16* __restrict__ W,
                                          const float* __restrict__ bias,
                                          _Float16* lds,  // 2 * 64*32 halves
                                          int m0, int N0, int tid, v8f acc[2][4]) {
  const int lane = tid & 31;
  const int ln = lane & 15, lh = lane >> 4;
  for (int t = 0; t < 4; t++) {
    float bb = bias[N0 + 16 * t + ln];
    for (int i = 0; i < 8; i++) { acc[0][t][i] = bb; acc[1][t][i] = bb; }
  }
  const _Float16* arow0 = xh + (size_t)(m0 + ln) * 512;
  const _Float16* arow1 = xh + (size_t)(m0 + 16 + ln) * 512;
  stage_W(W, lds, N0, 0, tid);
  stage_W(W, lds + 64 * 32, N0, 1, tid);
  v16h a0 = load_afrag(arow0, 0, lh);
  v16h a1 = load_afrag(arow1, 0, lh);
#pragma unroll 2
  for (int ks = 0; ks < 16; ks++) {
    WAIT_DS0();
    WAIT_ASYNCCNT(1);  // oldest stage (ks) complete; (ks+1) may be in flight
    BAR();
    const int ksn = (ks + 1 < 16) ? ks + 1 : 15;
    v16h a0n = load_afrag(arow0, 32 * ksn, lh);
    v16h a1n = load_afrag(arow1, 32 * ksn, lh);
    const _Float16* bb = lds + (ks & 1) * 64 * 32;
    v16h b[4];
    for (int t = 0; t < 4; t++) b[t] = load_bfrag_lds(bb + (16 * t + ln) * 32 + 16 * lh);
    for (int t = 0; t < 4; t++) {
      acc[0][t] = WMMA_F16(a0, b[t], acc[0][t]);
      acc[1][t] = WMMA_F16(a1, b[t], acc[1][t]);
    }
    WAIT_DS0();
    BAR();  // all waves done reading buf (ks&1)
    if (ks + 2 < 16) stage_W(W, lds + (ks & 1) * 64 * 32, N0, ks + 2, tid);
    a0 = a0n;
    a1 = a1n;
  }
}

// --------------------------- QKV projection GEMM ---------------------------
__global__ __launch_bounds__(256) void gemm_qkv_kernel(
    const _Float16* __restrict__ xh,
    const _Float16* __restrict__ Wq, const _Float16* __restrict__ Wk, const _Float16* __restrict__ Wv,
    const float* __restrict__ bq, const float* __restrict__ bk, const float* __restrict__ bv,
    _Float16* __restrict__ Qh, _Float16* __restrict__ Kh, _Float16* __restrict__ Vt) {
  __shared__ __align__(16) _Float16 sB[2 * 64 * 32];
  const int proj = blockIdx.z;
  const _Float16* W = (proj == 0) ? Wq : (proj == 1) ? Wk : Wv;
  const float* bias = (proj == 0) ? bq : (proj == 1) ? bk : bv;

  const int tid = threadIdx.x;
  const int lane = tid & 31, w = tid >> 5;
  const int ln = lane & 15, lh = lane >> 4;
  const int m0 = blockIdx.x * 256 + 32 * w;
  const int N0 = blockIdx.y * 64;

  v8f acc[2][4];
  gemm_tile(xh, W, bias, sB, m0, N0, tid, acc);

  for (int g = 0; g < 2; g++)
    for (int t = 0; t < 4; t++) {
      const int n = N0 + 16 * t + ln;
      const int h = n >> 6, d = n & 63;
      for (int i = 0; i < 8; i++) {
        const int row = m0 + 16 * g + i + 8 * lh;  // row = b*S + s
        const int b_ = row >> 11, s = row & 2047;
        const _Float16 v = (_Float16)acc[g][t][i];
        if (proj == 0)      Qh[((size_t)(b_ * 8 + h) * 2048 + s) * 64 + d] = v;
        else if (proj == 1) Kh[((size_t)(b_ * 8 + h) * 2048 + s) * 64 + d] = v;
        else                Vt[((size_t)(b_ * 8 + h) * 64 + d) * 2048 + s] = v;
      }
    }
}

// ------------------------ flash attention (masked) -------------------------
// Block: 128 threads (4 waves), one (b,h); wave w owns query rows q0..q0+15.
// K tile (32x64) and V tile (64x32, V transposed) are identical across the 4
// waves -> cooperatively staged in LDS, double-buffered one j-step ahead.
__device__ __forceinline__ void stage_kv(const _Float16* Kb, const _Float16* Vb,
                                         _Float16* sK, _Float16* sV, int j0, int tid) {
  {
    const int r = tid >> 2;  // 0..31
    for (int p = 0; p < 2; p++) {
      const int c = ((tid & 3) + 4 * p) * 8;  // halves
      async_copy16(Kb + (size_t)(j0 + r) * 64 + c, sK + r * 64 + c);
    }
  }
  {
    const int d = tid >> 1;  // 0..63
    for (int p = 0; p < 2; p++) {
      const int c = ((tid & 1) + 2 * p) * 8;  // halves
      async_copy16(Vb + (size_t)d * 2048 + j0 + c, sV + d * 32 + c);
    }
  }
}

__global__ __launch_bounds__(128) void attn_kernel(
    const _Float16* __restrict__ Qh, const _Float16* __restrict__ Kh,
    const _Float16* __restrict__ Vt, const int* __restrict__ ids,
    _Float16* __restrict__ AttnH) {
  __shared__ __align__(16) _Float16 sK[2 * 32 * 64];  // 8 KB
  __shared__ __align__(16) _Float16 sV[2 * 64 * 32];  // 8 KB
  __shared__ __align__(16) _Float16 sP[4 * 16 * 40];  // 5 KB (per-wave P transpose)

  const int tid = threadIdx.x;
  const int lane = tid & 31, w = tid >> 5;
  const int ln = lane & 15, lh = lane >> 4;
  const int bh = blockIdx.y;
  const int q0 = blockIdx.x * 64 + 16 * w;

  const _Float16* Qb = Qh + (size_t)bh * 2048 * 64;
  const _Float16* Kb = Kh + (size_t)bh * 2048 * 64;
  const _Float16* Vb = Vt + (size_t)bh * 64 * 2048;
  _Float16* ldsP = sP + w * 16 * 40;

  const _Float16* qrow = Qb + (size_t)(q0 + ln) * 64;
  const v16h aq0 = load_afrag(qrow, 0, lh);
  const v16h aq1 = load_afrag(qrow, 32, lh);

  int qid[8];
  for (int i = 0; i < 8; i++) qid[i] = ids[q0 + i + 8 * lh];  // C row = i + 8*lh

  float mrow[8], lrow[8];
  v8f acc[4];
  for (int i = 0; i < 8; i++) { mrow[i] = -1e30f; lrow[i] = 0.0f; }
  for (int t = 0; t < 4; t++)
    for (int i = 0; i < 8; i++) acc[t][i] = 0.0f;

  stage_kv(Kb, Vb, sK, sV, 0, tid);
  stage_kv(Kb, Vb, sK + 32 * 64, sV + 64 * 32, 32, tid);

#pragma unroll 2
  for (int jt = 0; jt < 64; jt++) {
    const int j0 = jt * 32;
    WAIT_DS0();
    WAIT_ASYNCCNT(4);  // oldest stage (4 async ops/thread) complete; next in flight
    BAR();
    const _Float16* tK = sK + (jt & 1) * 32 * 64;
    const _Float16* tV = sV + (jt & 1) * 64 * 32;

    // scores: 16x32 tile; batch all K fragment loads, then 4 wmma
    int kid[2];
    v16h kb[2][2];
    for (int t = 0; t < 2; t++) {
      kid[t] = ids[j0 + 16 * t + ln];
      kb[t][0] = load_bfrag_lds(tK + (16 * t + ln) * 64 + 16 * lh);
      kb[t][1] = load_bfrag_lds(tK + (16 * t + ln) * 64 + 32 + 16 * lh);
    }
    v8f sc[2];
    for (int t = 0; t < 2; t++) {
      v8f z;
      for (int i = 0; i < 8; i++) z[i] = 0.0f;
      z = WMMA_F16(aq0, kb[t][0], z);
      z = WMMA_F16(aq1, kb[t][1], z);
      sc[t] = z;
    }
    // scale + industry mask (branch-free; EXEC stays all-ones)
    for (int t = 0; t < 2; t++)
      for (int i = 0; i < 8; i++) {
        const float s = sc[t][i] * 0.125f;  // 1/sqrt(64)
        sc[t][i] = (kid[t] == qid[i]) ? s : -__builtin_inff();
      }
    // online softmax per row (rows replicated across 16-lane half)
    float p[2][8], alpha[8];
    for (int i = 0; i < 8; i++) {
      float vmax = fmaxf(sc[0][i], sc[1][i]);
      for (int off = 1; off < 16; off <<= 1) vmax = fmaxf(vmax, __shfl_xor(vmax, off));
      const float mnew = fmaxf(mrow[i], vmax);  // >= -1e30 always
      alpha[i] = __expf(mrow[i] - mnew);
      float ps = 0.0f;
      for (int t = 0; t < 2; t++) {
        p[t][i] = __expf(sc[t][i] - mnew);  // exp(-inf)=0 for masked
        ps += p[t][i];
      }
      for (int off = 1; off < 16; off <<= 1) ps += __shfl_xor(ps, off);
      lrow[i] = lrow[i] * alpha[i] + ps;
      mrow[i] = mnew;
    }
    for (int t = 0; t < 4; t++)
      for (int i = 0; i < 8; i++) acc[t][i] *= alpha[i];

    // P (C layout, f32) -> LDS -> A-fragment (f16, 16x32); per-wave region
    for (int t = 0; t < 2; t++)
      for (int i = 0; i < 8; i++)
        ldsP[(i + 8 * lh) * 40 + 16 * t + ln] = (_Float16)p[t][i];
    WAIT_DS0();
    v16h pa;
    ((uint4*)&pa)[0] = *(const uint4*)(ldsP + ln * 40 + 16 * lh);
    ((uint4*)&pa)[1] = *(const uint4*)(ldsP + ln * 40 + 16 * lh + 8);

    // O += P(16x32) x V(32x64); batch all V fragment loads, then 4 wmma
    v16h vb[4];
    for (int t = 0; t < 4; t++) vb[t] = load_bfrag_lds(tV + (16 * t + ln) * 32 + 16 * lh);
    for (int t = 0; t < 4; t++) acc[t] = WMMA_F16(pa, vb[t], acc[t]);

    WAIT_DS0();
    BAR();  // all waves done with buffer (jt&1)
    if (jt + 2 < 64) stage_kv(Kb, Vb, sK + (jt & 1) * 32 * 64, sV + (jt & 1) * 64 * 32,
                              (jt + 2) * 32, tid);
  }

  const int b_ = bh >> 3, h = bh & 7;
  for (int t = 0; t < 4; t++) {
    const int d = 16 * t + ln;
    for (int i = 0; i < 8; i++) {
      const int q = q0 + i + 8 * lh;
      AttnH[((size_t)(b_ * 2048 + q)) * 512 + h * 64 + d] = (_Float16)(acc[t][i] / lrow[i]);
    }
  }
}

// ----------------------------- output GEMM ---------------------------------
__global__ __launch_bounds__(256) void gemm_out_kernel(
    const _Float16* __restrict__ Ah, const _Float16* __restrict__ Wo,
    const float* __restrict__ bo, float* __restrict__ out) {
  __shared__ __align__(16) _Float16 sB[2 * 64 * 32];
  const int tid = threadIdx.x;
  const int lane = tid & 31, w = tid >> 5;
  const int ln = lane & 15, lh = lane >> 4;
  const int m0 = blockIdx.x * 256 + 32 * w;
  const int N0 = blockIdx.y * 64;

  v8f acc[2][4];
  gemm_tile(Ah, Wo, bo, sB, m0, N0, tid, acc);

  for (int g = 0; g < 2; g++)
    for (int t = 0; t < 4; t++) {
      const int n = N0 + 16 * t + ln;
      for (int i = 0; i < 8; i++) {
        const int row = m0 + 16 * g + i + 8 * lh;
        out[(size_t)row * 512 + n] = acc[g][t][i];
      }
    }
}

// ------------------------------- launcher ----------------------------------
extern "C" void kernel_launch(void* const* d_in, const int* in_sizes, int n_in,
                              void* d_out, int out_size, void* d_ws, size_t ws_size,
                              hipStream_t stream) {
  const float* x  = (const float*)d_in[0];
  const float* Wq = (const float*)d_in[1];
  const float* bq = (const float*)d_in[2];
  const float* Wk = (const float*)d_in[3];
  const float* bk = (const float*)d_in[4];
  const float* Wv = (const float*)d_in[5];
  const float* bv = (const float*)d_in[6];
  const float* Wo = (const float*)d_in[7];
  const float* bo = (const float*)d_in[8];
  const int*  ids = (const int*)d_in[9];
  float* out = (float*)d_out;

  char* ws = (char*)d_ws;
  const size_t XH_B = 16384ull * 512 * 2;  // 16 MB
  const size_t W_B  = 512ull * 512 * 2;    // 0.5 MB
  _Float16* xh  = (_Float16*)(ws);
  _Float16* Wqh = (_Float16*)(ws + XH_B);
  _Float16* Wkh = (_Float16*)(ws + XH_B + 1 * W_B);
  _Float16* Wvh = (_Float16*)(ws + XH_B + 2 * W_B);
  _Float16* Woh = (_Float16*)(ws + XH_B + 3 * W_B);
  char* p = ws + XH_B + 4 * W_B;
  _Float16* Qh    = (_Float16*)(p);             // [B*H, S, 64]
  _Float16* Kh    = (_Float16*)(p + 1 * XH_B);  // [B*H, S, 64]
  _Float16* Vt    = (_Float16*)(p + 2 * XH_B);  // [B*H, 64, S]
  _Float16* AttnH = (_Float16*)(p + 3 * XH_B);  // [B, S, E]

  const int nx = 8 * 2048 * 512, nw = 512 * 512;
  cvt_f32_f16<<<(nx + 255) / 256, 256, 0, stream>>>(x, xh, nx);
  cvt_f32_f16<<<(nw + 255) / 256, 256, 0, stream>>>(Wq, Wqh, nw);
  cvt_f32_f16<<<(nw + 255) / 256, 256, 0, stream>>>(Wk, Wkh, nw);
  cvt_f32_f16<<<(nw + 255) / 256, 256, 0, stream>>>(Wv, Wvh, nw);
  cvt_f32_f16<<<(nw + 255) / 256, 256, 0, stream>>>(Wo, Woh, nw);

  gemm_qkv_kernel<<<dim3(64, 8, 3), 256, 0, stream>>>(
      xh, Wqh, Wkh, Wvh, bq, bk, bv, Qh, Kh, Vt);

  attn_kernel<<<dim3(32, 64), 128, 0, stream>>>(Qh, Kh, Vt, ids, AttnH);

  gemm_out_kernel<<<dim3(64, 8), 256, 0, stream>>>(AttnH, Woh, bo, out);
}
